// Potential_17763984736960
// MI455X (gfx1250) — compile-verified, matrix operand order checked by reading, not run on hardware
//
#include <hip/hip_runtime.h>

typedef __attribute__((ext_vector_type(16))) _Float16 v16h;
typedef __attribute__((ext_vector_type(8)))  _Float16 v8h;
typedef __attribute__((ext_vector_type(8)))  float    v8f;

#define NUM_GRAPHS 8
#define NN 49152
#define NE 786432
#define KDIM 32
#define HDIM 64
#define NTILES (NE / 16)
#define E_OFF 0
#define F_OFF 8
#define S_OFF (8 + NN * 3)
#define H_OFF (S_OFF + 72)
#define OUT_TOT (H_OFF + 1)
#define NFRAG 24
#define EV_A3_TO_GPA 160.21766208f

// fast sigmoid: v_exp_f32 (TRANS pipe) + v_rcp_f32, no IEEE div chain
__device__ __forceinline__ float sigm_(float z) {
  return __builtin_amdgcn_rcpf(1.f + __expf(-z));
}

// A fragment: 16x32 f16 tile (ISA 7.12.2). Two aligned 16B LDS vector loads.
__device__ __forceinline__ v16h load_A_frag(const _Float16* p, int stride, int k0, int lane) {
  int m = lane & 15;
  int koff = (lane >> 4) << 3;
  const _Float16* r = p + m * stride + k0 + koff;
  v8h lo = *(const v8h*)(r);
  v8h hi = *(const v8h*)(r + 16);
  return __builtin_shufflevector(lo, hi, 0, 1, 2, 3, 4, 5, 6, 7, 8, 9, 10, 11, 12, 13, 14, 15);
}

#define WMMA(A, B, C) __builtin_amdgcn_wmma_f32_16x16x32_f16(false, (A), false, (B), (short)0, (C), false, false)

// Packed B-fragment ids:
//  0..3   W1   (fwd L1), k0=0,  n0=nt*16
//  4..11  W2   (fwd L2), kc*4+nt
// 12..19  W2^T (bwd dh1), kc*4+nt
// 20..23  W1^T (bwd drbf), kc*2+nt
__global__ __launch_bounds__(128) void edge_mlp_fused(
    const float* __restrict__ pos, const float* __restrict__ bond_vec,
    const int* __restrict__ src, const int* __restrict__ dst,
    const int* __restrict__ edge_graph, const float* __restrict__ centers,
    const float* __restrict__ W1, const float* __restrict__ b1,
    const float* __restrict__ W2, const float* __restrict__ b2,
    const float* __restrict__ W3, const float* __restrict__ b3,
    float* __restrict__ out_forces, float* __restrict__ ws_stress,
    float* __restrict__ ws_energy) {
  __shared__ __align__(16) _Float16 sBF[NFRAG * 32 * 16];  // pre-packed B fragments (24KB)
  __shared__ float sStress[NUM_GRAPHS * 9];
  __shared__ float sEnergy[NUM_GRAPHS];
  __shared__ __align__(16) _Float16 sA[4][16 * KDIM];      // rbf fwd / drbf bwd
  __shared__ __align__(16) _Float16 sT[4][16 * HDIM];      // a1 -> dz2 -> dz1

  // ---- pack all WMMA B fragments once per block ----
  for (int idx = threadIdx.x; idx < NFRAG * 32 * 16; idx += blockDim.x) {
    int frag = idx >> 9, rem = idx & 511;
    int ln = rem >> 4, h = rem & 15;
    int n = ln & 15, koff = (ln >> 4) << 3;
    int k = (h < 8) ? (koff + h) : (16 + koff + (h - 8));
    float v;
    if (frag < 4) {
      v = W1[k * HDIM + frag * 16 + n];
    } else if (frag < 12) {
      int f = frag - 4, kc = f >> 2, nt = f & 3;
      v = W2[(kc * 32 + k) * HDIM + nt * 16 + n];
    } else if (frag < 20) {
      int f = frag - 12, kc = f >> 2, nt = f & 3;        // W2^T: B[k][n] = W2[n][k]
      v = W2[(nt * 16 + n) * HDIM + kc * 32 + k];
    } else {
      int f = frag - 20, kc = f >> 1, nt = f & 1;        // W1^T: B[k][n] = W1[n][k]
      v = W1[(nt * 16 + n) * HDIM + kc * 32 + k];
    }
    sBF[idx] = (_Float16)v;
  }
  for (int i = threadIdx.x; i < NUM_GRAPHS * 9; i += blockDim.x) sStress[i] = 0.f;
  for (int i = threadIdx.x; i < NUM_GRAPHS; i += blockDim.x) sEnergy[i] = 0.f;
  __syncthreads();

  const int lane = threadIdx.x & 31;
  const int w = threadIdx.x >> 5;
  const int col = lane & 15;
  const int rb = (lane >> 4) << 3;
  const int kh = (lane >> 4) << 4;     // per-half k-range base for edge-scalar loops
  _Float16* A = sA[w];
  _Float16* T = sT[w];
  const v16h* BF = (const v16h*)sBF;

  // loop-invariant per-lane params in registers
  float b1r[4], b2r[4], w3r[4];
#pragma unroll
  for (int nt = 0; nt < 4; ++nt) {
    b1r[nt] = b1[nt * 16 + col];
    b2r[nt] = b2[nt * 16 + col];
    w3r[nt] = W3[nt * 16 + col];
  }
  const float b3r = b3[0];

  const int wpb = blockDim.x >> 5;
  const int gw = blockIdx.x * wpb + w;
  const int nw = gridDim.x * wpb;

  for (int tile = gw; tile < NTILES; tile += nw) {
    const int base = tile * 16;

    // ---- per-edge scalars: BOTH half-waves mirror edge (lane&15); k-work split by half ----
    const int e = base + col;
    const int eg = edge_graph[e];
    const int si = src[e];
    const int di = dst[e];
    const float bx = bond_vec[3 * e + 0], by = bond_vec[3 * e + 1], bz = bond_vec[3 * e + 2];
    const float dx = bx + pos[3 * di + 0] - pos[3 * si + 0];
    const float dy = by + pos[3 * di + 1] - pos[3 * si + 1];
    const float dz = bz + pos[3 * di + 2] - pos[3 * si + 2];
    const float d = sqrtf(dx * dx + dy * dy + dz * dz + 1e-12f);
#pragma unroll
    for (int kk = 0; kk < 16; kk += 8) {                  // this half's 16 of 32 k's
      v8h rv;
#pragma unroll
      for (int h = 0; h < 8; ++h) {
        float t = d - centers[kh + kk + h];               // vector loads, uniform per half-wave
        rv[h] = (_Float16)__expf(-t * t);
      }
      *(v8h*)(A + col * KDIM + kh + kk) = rv;
    }
    __builtin_amdgcn_wave_barrier();

    // ---- layer 1: z1 = rbf @ W1 + b1; a1 -> T; silu'(z1) -> registers ----
    v16h a0 = load_A_frag(A, KDIM, 0, lane);
    v8f c[4];
    float sp1[32];
#pragma unroll
    for (int nt = 0; nt < 4; ++nt) {
      v8f z = {};
      c[nt] = WMMA(a0, BF[nt * 32 + lane], z);
    }
#pragma unroll
    for (int nt = 0; nt < 4; ++nt)
#pragma unroll
      for (int r = 0; r < 8; ++r) {
        float z = c[nt][r] + b1r[nt];
        float sg = sigm_(z);
        float a1 = z * sg;
        T[(rb + r) * HDIM + nt * 16 + col] = (_Float16)a1;
        sp1[nt * 8 + r] = sg + a1 * (1.f - sg);           // silu'(z1)
      }
    __builtin_amdgcn_wave_barrier();

    // ---- layer 2: z2 = a1 @ W2 + b2; energy partial + dz2 straight from D regs ----
    v16h alo = load_A_frag(T, HDIM, 0, lane);
    v16h ahi = load_A_frag(T, HDIM, 32, lane);
#pragma unroll
    for (int nt = 0; nt < 4; ++nt) {
      v8f z = {};
      z = WMMA(alo, BF[(4 + nt) * 32 + lane], z);
      z = WMMA(ahi, BF[(8 + nt) * 32 + lane], z);
      c[nt] = z;
    }
    float ep = 0.f;
#pragma unroll
    for (int nt = 0; nt < 4; ++nt)
#pragma unroll
      for (int r = 0; r < 8; ++r) {
        float z = c[nt][r] + b2r[nt];
        float sg = sigm_(z);
        float a2 = z * sg;                                // silu(z2), shared
        ep += a2 * w3r[nt];
        T[(rb + r) * HDIM + nt * 16 + col] =
            (_Float16)(w3r[nt] * (sg + a2 * (1.f - sg))); // dz2 = W3*silu'(z2)
      }
    ep += __shfl_xor(ep, 16, 32);                         // fold upper half onto lower
    if (lane < 16) atomicAdd(&sEnergy[eg], ep + b3r);     // 16 lanes x b3 = 16*b3 per tile
    __builtin_amdgcn_wave_barrier();

    // ---- dh1 = dz2 @ W2^T ; dz1 = dh1 * silu'(z1) (from registers) ----
    alo = load_A_frag(T, HDIM, 0, lane);
    ahi = load_A_frag(T, HDIM, 32, lane);
#pragma unroll
    for (int nt = 0; nt < 4; ++nt) {
      v8f z = {};
      z = WMMA(alo, BF[(12 + nt) * 32 + lane], z);
      z = WMMA(ahi, BF[(16 + nt) * 32 + lane], z);
      c[nt] = z;
    }
#pragma unroll
    for (int nt = 0; nt < 4; ++nt)
#pragma unroll
      for (int r = 0; r < 8; ++r)
        T[(rb + r) * HDIM + nt * 16 + col] = (_Float16)(c[nt][r] * sp1[nt * 8 + r]);
    __builtin_amdgcn_wave_barrier();

    // ---- drbf = dz1 @ W1^T -> A ----
    alo = load_A_frag(T, HDIM, 0, lane);
    ahi = load_A_frag(T, HDIM, 32, lane);
#pragma unroll
    for (int nt = 0; nt < 2; ++nt) {
      v8f z = {};
      z = WMMA(alo, BF[(20 + nt) * 32 + lane], z);
      z = WMMA(ahi, BF[(22 + nt) * 32 + lane], z);
#pragma unroll
      for (int r = 0; r < 8; ++r)
        A[(rb + r) * KDIM + nt * 16 + col] = (_Float16)z[r];
    }
    __builtin_amdgcn_wave_barrier();

    // ---- dd (k-split across halves) -> ddisp -> split force scatter + stress ----
    float dd = 0.f;
#pragma unroll
    for (int kk = 0; kk < 16; kk += 8) {
      v8h dv = *(const v8h*)(A + col * KDIM + kh + kk);
#pragma unroll
      for (int h = 0; h < 8; ++h) {
        float t = d - centers[kh + kk + h];
        dd += (float)dv[h] * (-2.f * t) * __expf(-t * t);
      }
    }
    dd += __shfl_xor(dd, 16, 32);                         // both halves get full sum
    const float inv = dd * __builtin_amdgcn_rcpf(d);
    const float gx = inv * dx, gy = inv * dy, gz = inv * dz;  // dE/ddisp
    if (lane < 16) {                                      // lower half: dst scatter
      atomicAdd(&out_forces[3 * di + 0], -gx);
      atomicAdd(&out_forces[3 * di + 1], -gy);
      atomicAdd(&out_forces[3 * di + 2], -gz);
    } else {                                              // upper half: src scatter
      atomicAdd(&out_forces[3 * si + 0], gx);
      atomicAdd(&out_forces[3 * si + 1], gy);
      atomicAdd(&out_forces[3 * si + 2], gz);
    }
    if (lane < 16) {                                      // stress: f_ij = -gbv
      const float fx = -gx, fy = -gy, fz = -gz;
      float* S = &sStress[eg * 9];
      atomicAdd(&S[0], bx * fx); atomicAdd(&S[1], bx * fy); atomicAdd(&S[2], bx * fz);
      atomicAdd(&S[3], by * fx); atomicAdd(&S[4], by * fy); atomicAdd(&S[5], by * fz);
      atomicAdd(&S[6], bz * fx); atomicAdd(&S[7], bz * fy); atomicAdd(&S[8], bz * fz);
    }
  }

  __syncthreads();
  for (int i = threadIdx.x; i < NUM_GRAPHS * 9; i += blockDim.x) atomicAdd(&ws_stress[i], sStress[i]);
  for (int i = threadIdx.x; i < NUM_GRAPHS; i += blockDim.x) atomicAdd(&ws_energy[i], sEnergy[i]);
}

__global__ void zero_out(float* __restrict__ out, float* __restrict__ ws) {
  int i = blockIdx.x * blockDim.x + threadIdx.x;
  if (i < OUT_TOT) out[i] = 0.f;
  if (i < 80) ws[i] = 0.f;
}

__global__ void finalize(const float* __restrict__ ws_stress, const float* __restrict__ ws_energy,
                         const float* __restrict__ volume, float* __restrict__ out) {
  int i = threadIdx.x;
  if (i < NUM_GRAPHS) out[E_OFF + i] = ws_energy[i];
  if (i < NUM_GRAPHS * 9) {
    int g = i / 9;
    float v0 = volume[g * (NN / NUM_GRAPHS)];
    out[S_OFF + i] = -EV_A3_TO_GPA * ws_stress[i] / v0;
  }
  if (i == 0) out[H_OFF] = 0.f;
}

extern "C" void kernel_launch(void* const* d_in, const int* in_sizes, int n_in,
                              void* d_out, int out_size, void* d_ws, size_t ws_size,
                              hipStream_t stream) {
  (void)in_sizes; (void)n_in; (void)out_size; (void)ws_size;
  const float* pos      = (const float*)d_in[0];
  const float* bond_vec = (const float*)d_in[1];
  const int*   src      = (const int*)d_in[2];
  const int*   dst      = (const int*)d_in[3];
  const int*   eg       = (const int*)d_in[4];
  const float* volume   = (const float*)d_in[5];
  const float* centers  = (const float*)d_in[6];
  const float* W1 = (const float*)d_in[7];
  const float* b1 = (const float*)d_in[8];
  const float* W2 = (const float*)d_in[9];
  const float* b2 = (const float*)d_in[10];
  const float* W3 = (const float*)d_in[11];
  const float* b3 = (const float*)d_in[12];
  float* out = (float*)d_out;
  float* wsS = (float*)d_ws;   // [72] stress accum
  float* wsE = wsS + 72;       // [8]  energy accum

  zero_out<<<(OUT_TOT + 255) / 256, 256, 0, stream>>>(out, wsS);
  edge_mlp_fused<<<1024, 128, 0, stream>>>(pos, bond_vec, src, dst, eg, centers,
                                           W1, b1, W2, b2, W3, b3,
                                           out + F_OFF, wsS, wsE);
  finalize<<<1, 128, 0, stream>>>(wsS, wsE, volume, out);
}
